// GraphModel_1554778161790
// MI455X (gfx1250) — matrix operand order repaired and checked by reading, compiled, not run on hardware
//
#include <hip/hip_runtime.h>
#include <math.h>

// CDNA5 (gfx1250) GCN forward: two A@X GEMMs done with v_wmma_f32_16x16x32_bf16.
// A streams from HBM (512 MB total -> ~22us floor @ 23.3 TB/s); bf16 WMMA keeps
// the kernel memory-bound instead of FP32-VALU compute-bound.

typedef __attribute__((ext_vector_type(16))) __bf16 v16bf;
typedef __attribute__((ext_vector_type(8)))  float  v8f;

#define BATCH 64
#define NNODE 1024
#define CCH   50     // real channels
#define CPAD  64     // padded to 4 WMMA N-tiles
#define HDIM  512
#define KCLS  10
#define LPAD  264    // LDS row stride in bf16 elems (256 + 8 pad -> bank spread)

// ---------------------------------------------------------------------------
// XT[b][c][n] = bf16( x[b][n][0] * w1[0][c] ), zero for c >= 50 (transposed)
__global__ __launch_bounds__(256)
void k_xw1T(const float* __restrict__ x, const float* __restrict__ w1,
            __bf16* __restrict__ XT) {
    __shared__ float w1s[CCH];
    const int b = blockIdx.x >> 2;
    const int n = (blockIdx.x & 3) * 256 + threadIdx.x;
    if (threadIdx.x < CCH) w1s[threadIdx.x] = w1[threadIdx.x];
    __syncthreads();
    const float xv = x[(size_t)b * NNODE + n];
    __bf16* o = XT + (size_t)b * CPAD * NNODE + n;
    #pragma unroll 8
    for (int c = 0; c < CPAD; ++c)
        o[(size_t)c * NNODE] = (__bf16)(c < CCH ? xv * w1s[c] : 0.0f);
}

// ---------------------------------------------------------------------------
// XT[b][c][n] = bf16( sum_f H[b][n][f] * w2[f][c] ), zero for c >= 50
__global__ __launch_bounds__(256)
void k_xw2T(const float* __restrict__ H, const float* __restrict__ w2,
            __bf16* __restrict__ XT) {
    __shared__ float w2s[CCH * CCH];
    const int b = blockIdx.x >> 2;
    const int n = (blockIdx.x & 3) * 256 + threadIdx.x;
    for (int i = threadIdx.x; i < CCH * CCH; i += 256) w2s[i] = w2[i];
    __syncthreads();
    float h[CCH];
    const float* hr = H + ((size_t)b * NNODE + n) * CCH;
    #pragma unroll
    for (int f = 0; f < CCH; ++f) h[f] = hr[f];
    __bf16* o = XT + (size_t)b * CPAD * NNODE + n;
    for (int c = 0; c < CPAD; ++c) {
        float s = 0.0f;
        if (c < CCH) {
            #pragma unroll
            for (int f = 0; f < CCH; ++f) s += h[f] * w2s[f * CCH + c];
        }
        o[(size_t)c * NNODE] = (__bf16)s;
    }
}

// ---------------------------------------------------------------------------
// Out[b][m][c] = elu( sum_k A[b][m][k] * XT[b][c][k] + bias[c] ), c < 50.
// Grid: BATCH*8 blocks, 256 threads (8 waves). Each wave: one 16x64 tile.
// X tile double-buffered in LDS: one barrier per 256-K chunk; staging loads
// for chunk c+1 overlap the WMMA stream of chunk c.
__global__ __launch_bounds__(256)
void k_spmm(const float* __restrict__ A, const __bf16* __restrict__ XT,
            const float* __restrict__ bias, float* __restrict__ Out) {
    __shared__ __bf16 xlds[2][CPAD][LPAD];       // 2 x 33,792 B = 67,584 B

    const int b      = blockIdx.x >> 3;
    const int mchunk = blockIdx.x & 7;
    const int tid    = threadIdx.x;
    const int wave   = tid >> 5;
    const int lane   = tid & 31;
    const int l16    = lane & 15;
    const int khalf  = lane >> 4;                // wave32 half select
    const int mbase  = mchunk * 128 + wave * 16;

    const float*  arow = A  + ((size_t)b * NNODE + (mbase + l16)) * NNODE;
    const __bf16* xt   = XT + (size_t)b * CPAD * NNODE;

    v8f acc[4];
    #pragma unroll
    for (int j = 0; j < 4; ++j) acc[j] = (v8f){};

    // Stage 64 channels x 256 K of bf16 X into LDS (contiguous 16B copies).
    auto stage = [&](int chunk) {
        const int buf = chunk & 1;
        const int kb0 = chunk * 256;
        for (int i = tid; i < 2048; i += 256) {
            const int row = i >> 5, off = i & 31;
            const uint4 d = *reinterpret_cast<const uint4*>(
                xt + (size_t)row * NNODE + kb0 + off * 8);
            *reinterpret_cast<uint4*>(&xlds[buf][row][off * 8]) = d;
        }
    };

    stage(0);
    for (int chunk = 0; chunk < 4; ++chunk) {
        __syncthreads();                  // buf[chunk&1] staged; prev compute done
        if (chunk + 1 < 4) stage(chunk + 1);
        const int buf = chunk & 1;
        const int kb0 = chunk * 256;

        for (int kt = 0; kt < 256; kt += 32) {
            const int kg = kb0 + kt;
            // --- A fragment: 16x32 bf16, per ISA 16-bit A layout ---
            // lanes 0-15 (khalf=0): K = {0..7, 16..23}; lanes 16-31: {8..15, 24..31}
            const float* ap = arow + kg + khalf * 8;
            const float4 f0 = reinterpret_cast<const float4*>(ap)[0];
            const float4 f1 = reinterpret_cast<const float4*>(ap)[1];
            const float4 f2 = reinterpret_cast<const float4*>(ap + 16)[0];
            const float4 f3 = reinterpret_cast<const float4*>(ap + 16)[1];
            v16bf af;
            af[0]  = (__bf16)f0.x; af[1]  = (__bf16)f0.y;
            af[2]  = (__bf16)f0.z; af[3]  = (__bf16)f0.w;
            af[4]  = (__bf16)f1.x; af[5]  = (__bf16)f1.y;
            af[6]  = (__bf16)f1.z; af[7]  = (__bf16)f1.w;
            af[8]  = (__bf16)f2.x; af[9]  = (__bf16)f2.y;
            af[10] = (__bf16)f2.z; af[11] = (__bf16)f2.w;
            af[12] = (__bf16)f3.x; af[13] = (__bf16)f3.y;
            af[14] = (__bf16)f3.z; af[15] = (__bf16)f3.w;

            // Near-scope prefetch 1KB (one chunk) ahead on this row's stream.
            if (kg + 256 < NNODE)
                __builtin_prefetch(arow + kg + 256 + khalf * 8, 0, 3);

            // --- B fragments: 32x16 bf16; lane = column, K = {0..7,16..23} / {8..15,24..31} ---
            const int klb = kt + khalf * 8;
            #pragma unroll
            for (int j = 0; j < 4; ++j) {
                const __bf16* bp = &xlds[buf][j * 16 + l16][klb];
                union { uint4 u[2]; v16bf v; } bb;
                bb.u[0] = *reinterpret_cast<const uint4*>(bp);
                bb.u[1] = *reinterpret_cast<const uint4*>(bp + 16);
                acc[j] = __builtin_amdgcn_wmma_f32_16x16x32_bf16(
                    false, af, false, bb.v, (short)0, acc[j], false, false);
            }
        }
    }

    // Epilogue: C/D layout -> M = khalf*8 + v, N = l16. Bias + ELU, drop pad cols.
    #pragma unroll
    for (int j = 0; j < 4; ++j) {
        const int col = j * 16 + l16;
        if (col < CCH) {
            const float bc = bias[col];
            #pragma unroll
            for (int v = 0; v < 8; ++v) {
                const int row = mbase + khalf * 8 + v;
                float s = acc[j][v] + bc;
                Out[((size_t)b * NNODE + row) * CCH + col] =
                    (s > 0.0f) ? s : (expf(s) - 1.0f);
            }
        }
    }
}

// ---------------------------------------------------------------------------
// pooled[b][c] = sum_n H[b][n][c]
__global__ __launch_bounds__(256)
void k_pool(const float* __restrict__ H, float* __restrict__ P) {
    __shared__ float ps[4][64];
    const int b = blockIdx.x;
    const int c = threadIdx.x & 63;
    const int seg = threadIdx.x >> 6;
    float s = 0.0f;
    if (c < CCH)
        for (int n = seg * 256; n < (seg + 1) * 256; ++n)
            s += H[((size_t)b * NNODE + n) * CCH + c];
    ps[seg][c] = s;
    __syncthreads();
    if (threadIdx.x < CCH)
        P[b * CCH + threadIdx.x] = ps[0][threadIdx.x] + ps[1][threadIdx.x] +
                                   ps[2][threadIdx.x] + ps[3][threadIdx.x];
}

// ---------------------------------------------------------------------------
// out[b] = softmax( relu(pooled[b] @ wf1 + bf1) @ wf2 + bf2 )
__global__ __launch_bounds__(512)
void k_head(const float* __restrict__ P, const float* __restrict__ wf1,
            const float* __restrict__ bf1, const float* __restrict__ wf2,
            const float* __restrict__ bf2, float* __restrict__ out) {
    __shared__ float pl[CCH];
    __shared__ float hb[HDIM];
    __shared__ float lg[KCLS];
    const int b = blockIdx.x, t = threadIdx.x;
    if (t < CCH) pl[t] = P[b * CCH + t];
    __syncthreads();
    float s = bf1[t];
    #pragma unroll
    for (int c = 0; c < CCH; ++c) s += pl[c] * wf1[c * HDIM + t];
    hb[t] = s > 0.0f ? s : 0.0f;
    __syncthreads();
    if (t < KCLS) {
        float l = bf2[t];
        for (int j = 0; j < HDIM; ++j) l += hb[j] * wf2[j * KCLS + t];
        lg[t] = l;
    }
    __syncthreads();
    if (t == 0) {
        float mx = lg[0];
        for (int k = 1; k < KCLS; ++k) mx = fmaxf(mx, lg[k]);
        float e[KCLS], sm = 0.0f;
        for (int k = 0; k < KCLS; ++k) { e[k] = expf(lg[k] - mx); sm += e[k]; }
        for (int k = 0; k < KCLS; ++k) out[b * KCLS + k] = e[k] / sm;
    }
}

// ---------------------------------------------------------------------------
extern "C" void kernel_launch(void* const* d_in, const int* in_sizes, int n_in,
                              void* d_out, int out_size, void* d_ws, size_t ws_size,
                              hipStream_t stream) {
    const float* x   = (const float*)d_in[0];
    const float* a   = (const float*)d_in[1];
    const float* w1  = (const float*)d_in[2];
    const float* b1  = (const float*)d_in[3];
    const float* w2  = (const float*)d_in[4];
    const float* b2  = (const float*)d_in[5];
    const float* wf1 = (const float*)d_in[6];
    const float* bf1 = (const float*)d_in[7];
    const float* wf2 = (const float*)d_in[8];
    const float* bf2 = (const float*)d_in[9];
    float* out = (float*)d_out;

    char* ws = (char*)d_ws;
    const size_t xt_bytes = (size_t)BATCH * CPAD * NNODE * sizeof(__bf16); // 8 MB
    const size_t h_bytes  = (size_t)BATCH * NNODE * CCH * sizeof(float);   // 13.1 MB
    __bf16* XT = (__bf16*)ws;                       // reused by both layers
    float*  H  = (float*)(ws + xt_bytes);           // h1, then overwritten by h2
    float*  P  = (float*)(ws + xt_bytes + h_bytes); // pooled [64,50]

    k_xw1T<<<BATCH * 4, 256, 0, stream>>>(x, w1, XT);
    k_spmm<<<BATCH * 8, 256, 0, stream>>>(a, XT, b1, H);   // layer 1 (WMMA)
    k_xw2T<<<BATCH * 4, 256, 0, stream>>>(H, w2, XT);
    k_spmm<<<BATCH * 8, 256, 0, stream>>>(a, XT, b2, H);   // layer 2 (WMMA)
    k_pool<<<BATCH, 256, 0, stream>>>(H, P);
    k_head<<<BATCH, 512, 0, stream>>>(P, wf1, bf1, wf2, bf2, out);
}
